// CDSA_20847771254992
// MI455X (gfx1250) — matrix operand order repaired and checked by reading, not compiled, not run on hardware
//
#include <hip/hip_runtime.h>
#include <hip/hip_bf16.h>
#include <cmath>

// Workspace: w_f16 [8][9216][64] = 9,437,184 B + off f32 [8][18][9216] = 5,308,416 B
//          = 14,745,600 bytes total.

typedef __attribute__((ext_vector_type(16))) _Float16 v16h;
typedef __attribute__((ext_vector_type(8)))  _Float16 v8h;
typedef __attribute__((ext_vector_type(8)))  float    v8f;

#define BATCH 8
#define CIN   256
#define CM    64
#define HH    96
#define WW    96
#define PIX   (HH*WW)   // 9216

// --- WMMA helpers -----------------------------------------------------------
// A operand (16x32 f16, MxK): lane m=lane%16, g=lane/16.
//   VGPR0-7 hold K = Kb + 16g + 0..15 (16 contiguous K per lane)
// B operand (32x16 f16, KxN): lane n=lane%16, g=lane/16.
//   VGPR v holds K = Kb + 16g + 2v,2v+1  -> 16 contiguous K per lane.
// C/D (16x16 f32): VGPR r holds M = r + 8g, N = lane%16.
__device__ __forceinline__ v16h ld_pair(const _Float16* p0, const _Float16* p1) {
  v8h a = *(const v8h*)p0;
  v8h b = *(const v8h*)p1;
  return __builtin_shufflevector(a, b, 0,1,2,3,4,5,6,7,8,9,10,11,12,13,14,15);
}
__device__ __forceinline__ v16h ldA(const _Float16* p) { return ld_pair(p, p + 8); }
__device__ __forceinline__ v16h ldB(const _Float16* p) { return ld_pair(p, p + 8); }
__device__ __forceinline__ v8f wmma_f16(v16h a, v16h b, v8f c) {
  return __builtin_amdgcn_wmma_f32_16x16x32_f16(false, a, false, b, (short)0, c, false, false);
}

// CDNA5 async copy global->LDS (ASYNCcnt-tracked). lds_off is the byte offset
// inside the workgroup's (dynamic) LDS allocation; each lane supplies its own
// LDS destination and global source.
__device__ __forceinline__ void async_ld_b128(unsigned lds_off, const void* gptr) {
  asm volatile("global_load_async_to_lds_b128 %0, %1, off"
               :: "v"(lds_off), "v"(gptr) : "memory");
}
__device__ __forceinline__ void wait_async0() {
  asm volatile("s_wait_asynccnt 0x0" ::: "memory");
}

// ============================================================================
// K1: pwconv1.  wh[b,p,o] = sum_c x[b,c,p]*w1[o,c] + b1[o]   (f16, ch-contig)
// GEMM M=64, K=256, N=73728.  WG = 32 pixels of one batch, 8 waves.
// LDS: w1 f16 [64][272], x tile f16 [32 px][272 (K-contig)]
// ============================================================================
__global__ __launch_bounds__(256) void k1_pw1(
    const float* __restrict__ x, const float* __restrict__ w1,
    const float* __restrict__ b1, _Float16* __restrict__ wh) {
  extern __shared__ __align__(16) char smem[];
  _Float16* w1L = (_Float16*)smem;        // 64*272
  _Float16* xt  = w1L + 64 * 272;         // 32*272
  const int tid = threadIdx.x;
  const int bwg = blockIdx.x / 288;
  const int p0  = (blockIdx.x % 288) * 32;

  for (int i = tid; i < 64 * 256; i += 256) {
    int o = i >> 8, c = i & 255;
    w1L[o * 272 + c] = (_Float16)w1[i];
  }
  for (int i = tid; i < 256 * 32; i += 256) {
    int c = i >> 5, px = i & 31;
    xt[px * 272 + c] = (_Float16)x[(size_t)(bwg * 256 + c) * PIX + p0 + px];
  }
  __syncthreads();

  const int wid = tid >> 5, lane = tid & 31, l16 = lane & 15, g = lane >> 4;
  const int mt = wid & 3, nt = wid >> 2;
  v8f acc = {};
#pragma unroll
  for (int k0 = 0; k0 < 256; k0 += 32) {
    v16h a  = ldA(w1L + (mt * 16 + l16) * 272 + k0 + g * 16);
    v16h bb = ldB(xt + (nt * 16 + l16) * 272 + k0 + g * 16);
    acc = wmma_f16(a, bb, acc);
  }
#pragma unroll
  for (int r = 0; r < 8; ++r) {
    int ch = mt * 16 + r + 8 * g;
    wh[((size_t)bwg * PIX + p0 + nt * 16 + l16) * 64 + ch] =
        (_Float16)(acc[r] + b1[ch]);
  }
}

// ============================================================================
// K2: offset conv (3x3, SAME, 18 out ch) via WMMA.
// WG = 16x16 pixel tile. LDS: w halo tile f16 [18][18][72] (async-staged from
// wh, zero-padded at image border), woff f16 [32][584] (rows 18..31 zero).
// K-dim = k*64+c (576).
// ============================================================================
__global__ __launch_bounds__(256) void k2_off(
    const _Float16* __restrict__ wh, const float* __restrict__ woff,
    const float* __restrict__ boff, float* __restrict__ offb) {
  extern __shared__ __align__(16) char smem[];
  _Float16* wt = (_Float16*)smem;         // [18*18][72], smem offset 0
  _Float16* wA = wt + 18 * 18 * 72;       // 32*584
  const int tid = threadIdx.x;
  const int bwg = blockIdx.x / 36;
  const int tt0 = blockIdx.x % 36;
  const int y0 = (tt0 / 6) * 16, x0 = (tt0 % 6) * 16;

  // async byte-copy of the halo tile: one pixel = 64 f16 = 8 x b128
  for (int i = tid; i < 18 * 18; i += 256) {
    int xx = i % 18, yy = i / 18;
    int gy = y0 - 1 + yy, gx = x0 - 1 + xx;
    unsigned lb = (unsigned)(i * 72 * 2);
    if ((unsigned)gy < (unsigned)HH && (unsigned)gx < (unsigned)WW) {
      const _Float16* src = wh + ((size_t)bwg * PIX + gy * WW + gx) * 64;
#pragma unroll
      for (int ck = 0; ck < 8; ++ck) async_ld_b128(lb + ck * 16, src + ck * 8);
    } else {
      v8h z = {};
#pragma unroll
      for (int ck = 0; ck < 8; ++ck) *(v8h*)(wt + i * 72 + ck * 8) = z;
    }
  }
  for (int i = tid; i < 32 * 576; i += 256) {
    int row = i / 576, col = i % 576;
    int c = col & 63, k = col >> 6;
    float v = (row < 18) ? woff[(row * 64 + c) * 9 + k] : 0.f;
    wA[row * 584 + col] = (_Float16)v;
  }
  wait_async0();
  __syncthreads();

  const int wid = tid >> 5, lane = tid & 31, l16 = lane & 15, g = lane >> 4;
  for (int hh = 0; hh < 2; ++hh) {
    const int nt = wid + hh * 8;                 // tile row = N-tile
    v8f a0 = {}, a1 = {};
#pragma unroll
    for (int k = 0; k < 9; ++k) {
      const int sy = nt + (k / 3);               // halo origin is -1
      const int sx = l16 + (k % 3);
#pragma unroll
      for (int cc = 0; cc < 64; cc += 32) {
        v16h bb  = ldB(wt + (sy * 18 + sx) * 72 + cc + g * 16);
        v16h aa0 = ldA(wA + l16 * 584 + k * 64 + cc + g * 16);
        a0 = wmma_f16(aa0, bb, a0);
        v16h aa1 = ldA(wA + (16 + l16) * 584 + k * 64 + cc + g * 16);
        a1 = wmma_f16(aa1, bb, a1);
      }
    }
    const int gp = (y0 + nt) * WW + x0 + l16;
#pragma unroll
    for (int r = 0; r < 8; ++r) {
      int o = r + 8 * g;
      offb[(size_t)(bwg * 18 + o) * PIX + gp] = a0[r] + boff[o];
    }
    if (g == 0) {
#pragma unroll
      for (int r = 0; r < 2; ++r) {
        int o = 16 + r;
        offb[(size_t)(bwg * 18 + o) * PIX + gp] = a1[r] + boff[o];
      }
    }
  }
}

// ============================================================================
// K3: deformable conv (bilinear + einsum) + BN + pwconv2 + sigmoid gate.
// WG = 16x16 pixel tile of one batch.
// LDS: w halo tile f16 [32][32][72] (halo +-8, async-staged; taps outside the
//      tile fall back to b128 gathers from L2-resident wh),
//      wd f16 [64][584], w2 f16 [256][72], ybn staging [8 waves][16 px][72].
// ============================================================================
__global__ __launch_bounds__(256) void k3_deform(
    const _Float16* __restrict__ wh, const float* __restrict__ offb,
    const float* __restrict__ wd, const float* __restrict__ bd,
    const float* __restrict__ gamma, const float* __restrict__ beta,
    const float* __restrict__ rmean, const float* __restrict__ rvar,
    const float* __restrict__ w2, const float* __restrict__ b2,
    const float* __restrict__ x, float* __restrict__ out) {
  extern __shared__ __align__(16) char smem[];
  _Float16* wt  = (_Float16*)smem;         // [32*32][72], smem offset 0
  _Float16* wdL = wt + 32 * 32 * 72;       // 64*584
  _Float16* w2L = wdL + 64 * 584;          // 256*72
  _Float16* ybn = w2L + 256 * 72;          // 8*16*72
  const int tid = threadIdx.x;
  const int bwg = blockIdx.x / 36;
  const int tt0 = blockIdx.x % 36;
  const int y0 = (tt0 / 6) * 16, x0 = (tt0 % 6) * 16;

  for (int i = tid; i < 32 * 32; i += 256) {
    int xx = i & 31, yy = i >> 5;
    int gy = y0 - 8 + yy, gx = x0 - 8 + xx;
    unsigned lb = (unsigned)(i * 72 * 2);
    if ((unsigned)gy < (unsigned)HH && (unsigned)gx < (unsigned)WW) {
      const _Float16* src = wh + ((size_t)bwg * PIX + gy * WW + gx) * 64;
#pragma unroll
      for (int ck = 0; ck < 8; ++ck) async_ld_b128(lb + ck * 16, src + ck * 8);
    } else {
      v8h z = {};
#pragma unroll
      for (int ck = 0; ck < 8; ++ck) *(v8h*)(wt + i * 72 + ck * 8) = z;
    }
  }
  for (int i = tid; i < 64 * 576; i += 256) {
    int row = i / 576, col = i % 576;
    int c = col & 63, k = col >> 6;
    wdL[row * 584 + col] = (_Float16)wd[(row * 64 + c) * 9 + k];
  }
  for (int i = tid; i < 256 * 64; i += 256) {
    int row = i >> 6, c = i & 63;
    w2L[row * 72 + c] = (_Float16)w2[i];
  }
  wait_async0();
  __syncthreads();

  const int wid = tid >> 5, lane = tid & 31, l16 = lane & 15, g = lane >> 4;
  _Float16* myY = ybn + wid * 16 * 72;

  for (int hh = 0; hh < 2; ++hh) {
    const int nt = wid + hh * 8;
    const int gy = y0 + nt, gx = x0 + l16;
    const int gp = gy * WW + gx;

    // warm the gate-stage x lines (global_prefetch) while we do the deform math
    const size_t gpr = (size_t)gy * WW + x0;
#pragma unroll
    for (int q = 0; q < 8; ++q)
      __builtin_prefetch(&x[(size_t)(bwg * 256 + q * 32 + lane) * PIX + gpr], 0, 1);

    v8f acc[4] = {v8f{}, v8f{}, v8f{}, v8f{}};

    for (int k = 0; k < 9; ++k) {
      const float offy = offb[(size_t)(bwg * 18 + 2 * k)     * PIX + gp];
      const float offx = offb[(size_t)(bwg * 18 + 2 * k + 1) * PIX + gp];
      const float py = (float)(gy + k / 3 - 1) + offy;
      const float px = (float)(gx + k % 3 - 1) + offx;
      const float fpy = floorf(py), fpx = floorf(px);
      const float fy = py - fpy, fx = px - fpx;
      const int iy0 = (int)fpy, ix0 = (int)fpx;
      float wgt[4]; int tyc[4], txc[4];
#pragma unroll
      for (int t = 0; t < 4; ++t) {
        int dy = t >> 1, dx = t & 1;
        int yc = iy0 + dy, xc = ix0 + dx;
        float w = (dy ? fy : 1.f - fy) * (dx ? fx : 1.f - fx);
        if (yc < 0 || yc > HH - 1 || xc < 0 || xc > WW - 1) w = 0.f;  // zero pad
        wgt[t] = w; tyc[t] = yc; txc[t] = xc;
      }
#pragma unroll
      for (int cc = 0; cc < 64; cc += 32) {
        const int cbase = cc + g * 16;
        float s[16];
#pragma unroll
        for (int i = 0; i < 16; ++i) s[i] = 0.f;
#pragma unroll
        for (int t = 0; t < 4; ++t) {
          if (wgt[t] == 0.f) continue;
          const int ly = tyc[t] - (y0 - 8), lx = txc[t] - (x0 - 8);
          if ((unsigned)ly < 32u && (unsigned)lx < 32u) {
            v16h tv = ldB(wt + (ly * 32 + lx) * 72 + cbase);
#pragma unroll
            for (int i = 0; i < 16; ++i) s[i] += wgt[t] * (float)tv[i];
          } else {  // rare large-offset fallback: wh is L2-resident, ch-contig
            v16h tv = ldB(wh + ((size_t)bwg * PIX + tyc[t] * WW + txc[t]) * 64 + cbase);
#pragma unroll
            for (int i = 0; i < 16; ++i) s[i] += wgt[t] * (float)tv[i];
          }
        }
        v16h bv;
#pragma unroll
        for (int i = 0; i < 16; ++i) bv[i] = (_Float16)s[i];
#pragma unroll
        for (int mt = 0; mt < 4; ++mt) {
          v16h aa = ldA(wdL + (mt * 16 + l16) * 584 + k * 64 + cc + g * 16);
          acc[mt] = wmma_f16(aa, bv, acc[mt]);
        }
      }
    }

    // BN (eval) with bd folded in, stash f16 in per-wave LDS area [px][c]
#pragma unroll
    for (int mt = 0; mt < 4; ++mt) {
#pragma unroll
      for (int r = 0; r < 8; ++r) {
        int c = mt * 16 + r + 8 * g;
        float inv = gamma[c] * rsqrtf(rvar[c] + 1e-5f);
        float sh  = (bd[c] - rmean[c]) * inv + beta[c];
        myY[l16 * 72 + c] = (_Float16)(acc[mt][r] * inv + sh);
      }
    }
    // same-wave DS ops are in-order; fence the compiler before re-reading
    asm volatile("s_wait_dscnt 0" ::: "memory");

    v16h bv0 = ldB(myY + l16 * 72 + 0  + g * 16);
    v16h bv1 = ldB(myY + l16 * 72 + 32 + g * 16);
    for (int mt2 = 0; mt2 < 16; ++mt2) {
      v8f a2 = {};
      v16h aa0 = ldA(w2L + (mt2 * 16 + l16) * 72 + 0  + g * 16);
      a2 = wmma_f16(aa0, bv0, a2);
      v16h aa1 = ldA(w2L + (mt2 * 16 + l16) * 72 + 32 + g * 16);
      a2 = wmma_f16(aa1, bv1, a2);
#pragma unroll
      for (int r = 0; r < 8; ++r) {
        int o2 = mt2 * 16 + r + 8 * g;
        size_t idx = (size_t)(bwg * 256 + o2) * PIX + gp;
        float v = a2[r] + b2[o2];
        out[idx] = x[idx] * (1.f / (1.f + __expf(-v)));
      }
    }
  }
}

// ============================================================================
extern "C" void kernel_launch(void* const* d_in, const int* in_sizes, int n_in,
                              void* d_out, int out_size, void* d_ws, size_t ws_size,
                              hipStream_t stream) {
  (void)in_sizes; (void)n_in; (void)out_size; (void)ws_size;
  const float* x     = (const float*)d_in[0];
  const float* w1    = (const float*)d_in[1];
  const float* b1    = (const float*)d_in[2];
  const float* woff  = (const float*)d_in[3];
  const float* boff  = (const float*)d_in[4];
  const float* wd    = (const float*)d_in[5];
  const float* bd    = (const float*)d_in[6];
  const float* gamma = (const float*)d_in[7];
  const float* beta  = (const float*)d_in[8];
  const float* rmean = (const float*)d_in[9];
  const float* rvar  = (const float*)d_in[10];
  const float* w2    = (const float*)d_in[11];
  const float* b2    = (const float*)d_in[12];

  _Float16* wh = (_Float16*)d_ws;                               // [8][9216][64] f16
  float* offb  = (float*)((char*)d_ws + (size_t)BATCH * PIX * 64 * 2); // [8][18][9216] f32

  const size_t shm1 = (size_t)(64 * 272 + 32 * 272) * 2;                    // 52,224
  const size_t shm2 = (size_t)(18 * 18 * 72 + 32 * 584) * 2;                // 84,032
  const size_t shm3 = (size_t)(32 * 32 * 72 + 64 * 584 + 256 * 72 + 8 * 16 * 72) * 2; // 277,504

  // allow >64KB dynamic LDS (WGP has 320KB); idempotent, not a stream op
  (void)hipFuncSetAttribute((const void*)k2_off,
      hipFuncAttributeMaxDynamicSharedMemorySize, (int)shm2);
  (void)hipFuncSetAttribute((const void*)k3_deform,
      hipFuncAttributeMaxDynamicSharedMemorySize, (int)shm3);

  k1_pw1<<<dim3(BATCH * 288), dim3(256), shm1, stream>>>(x, w1, b1, wh);
  k2_off<<<dim3(BATCH * 36), dim3(256), shm2, stream>>>(wh, woff, boff, offb);
  k3_deform<<<dim3(BATCH * 36), dim3(256), shm3, stream>>>(
      wh, offb, wd, bd, gamma, beta, rmean, rvar, w2, b2, x, (float*)d_out);
}
